// _HSA_prefill_15547781612183
// MI455X (gfx1250) — matrix operand order repaired and checked by reading, not compiled, not run on hardware
//
#include <hip/hip_runtime.h>

typedef __attribute__((ext_vector_type(16))) _Float16 v16h;
typedef __attribute__((ext_vector_type(8)))  _Float16 v8h;
typedef __attribute__((ext_vector_type(8)))  float    v8f;
typedef __attribute__((ext_vector_type(4)))  float    v4f;

#define LQ   256
#define HQn  32
#define HKV  2
#define DH   128
#define SBLK 16
#define BSZ  64
#define Gg   16   // query heads per kv head

// One workgroup per (l, h): 4 waves x 32 lanes. Each wave processes 4 of the
// 16 sparse KV blocks (per-block softmax is independent), accumulating a
// private 16x128 f32 output tile in WMMA accumulators; partials are reduced
// across waves through LDS at the end.
__global__ __launch_bounds__(128) void sparse_blk_attn_kernel(
    const float* __restrict__ q, const float* __restrict__ k,
    const float* __restrict__ v, const float* __restrict__ w,
    const int* __restrict__ bidx, float* __restrict__ out)
{
    __shared__ __align__(16) float     qs[Gg * DH];        // 8 KB: Q tile f32
    __shared__ __align__(16) _Float16  ps[4][Gg * BSZ];    // 8 KB: P (per wave)
    __shared__ __align__(16) float     op[4][Gg * DH];     // 32 KB: O partials

    const int bid  = blockIdx.x;
    const int l    = bid >> 1;
    const int h    = bid & 1;
    const int tid  = threadIdx.x;
    const int wave = tid >> 5;
    const int lane = tid & 31;
    const int half_ = lane >> 4;     // which 16-lane group
    const int l16   = lane & 15;

    // ---- stage the Q tile (16 heads x 128 dims, 8 KB contiguous) into LDS
    //      with gfx1250 async global->LDS copies (no VGPR round trip) ----
    const int qbase = (l * HQn + h * Gg) * DH;   // 2048 contiguous floats
    {
        const float* gsrc = q + qbase;
#pragma unroll
        for (int i = 0; i < 4; ++i) {
            const int elt = (tid + i * 128) * 4;           // float4 granule
            unsigned lds_off = (unsigned)(uintptr_t)(&qs[elt]);
            unsigned long long ga = (unsigned long long)(uintptr_t)(gsrc + elt);
            asm volatile("global_load_async_to_lds_b128 %0, %1, off"
                         :: "v"(lds_off), "v"(ga) : "memory");
        }
        asm volatile("s_wait_asynccnt 0" ::: "memory");
    }
    __syncthreads();

    // ---- build Q A-fragments (M=head, contraction=d), 4 k-steps of 32 ----
    // A k order used here (contiguous, B matches): element e -> k = half*16 + e
    v16h qa[4];
#pragma unroll
    for (int t = 0; t < 4; ++t) {
        const float* row = qs + l16 * DH + t * 32 + half_ * 16;
#pragma unroll
        for (int i = 0; i < 8; ++i) {
            qa[t][i]     = (_Float16)row[i];        // k = 32t + half*16 + i
            qa[t][8 + i] = (_Float16)row[8 + i];    // k = 32t + half*16 + 8 + i
        }
    }

    v8f od[8];
#pragma unroll
    for (int jd = 0; jd < 8; ++jd)
#pragma unroll
        for (int vv = 0; vv < 8; ++vv) od[jd][vv] = 0.0f;

    const float scale = 0.08838834764831845f;   // 1/sqrt(128)
    _Float16* psw = &ps[wave][0];

    for (int si = 0; si < 4; ++si) {
        const int s  = wave * 4 + si;
        const int bi = bidx[(l * HKV + h) * SBLK + s];

        // ---- prefetch the next sparse block's K/V rows (data-dependent
        //      gather: overlap the pointer chase with this block's WMMAs) ----
        if (si < 3) {
            const int bi2 = bidx[(l * HKV + h) * SBLK + s + 1];
            if (bi2 >= 0) {
                const long nb = (long)bi2 * BSZ * (HKV * DH) + h * DH;
#pragma unroll
                for (int r = 0; r < 4; ++r) {
                    __builtin_prefetch(k + nb + (long)(r * 16 + l16) * (HKV * DH), 0, 1);
                    __builtin_prefetch(v + nb + (long)(r * 16 + l16) * (HKV * DH), 0, 1);
                }
            }
        }

        if (bi < 0) continue;   // uniform across the wave: EXEC stays all-1s

        const long kvbase = (long)bi * BSZ * (HKV * DH) + h * DH;

        // ---- scores: c[j] (16 heads x 16 keys), j = key tile within block ----
        v8f c[4];
#pragma unroll
        for (int j = 0; j < 4; ++j)
#pragma unroll
            for (int vv = 0; vv < 8; ++vv) c[j][vv] = 0.0f;

#pragma unroll
        for (int j = 0; j < 4; ++j) {
            const float* krow = k + kvbase + (long)(16 * j + l16) * (HKV * DH);
#pragma unroll
            for (int t = 0; t < 4; ++t) {
                // B layout: lane -> N=key, element e -> contraction d = 32t + half*16 + e
                const float* kp = krow + t * 32 + half_ * 16;
                v16h kb;
#pragma unroll
                for (int c4 = 0; c4 < 4; ++c4) {
                    v4f f = *reinterpret_cast<const v4f*>(kp + c4 * 4);
#pragma unroll
                    for (int x = 0; x < 4; ++x) kb[c4 * 4 + x] = (_Float16)f[x];
                }
                c[j] = __builtin_amdgcn_wmma_f32_16x16x32_f16(
                    false, qa[t], false, kb, (short)0, c[j], false, false);
            }
        }

        // ---- per-block softmax over the 64 keys, then * (w / sum) ----
        // C layout: c[j][vv] holds (M = vv + half*8, key = 16j + l16)
#pragma unroll
        for (int vv = 0; vv < 8; ++vv) {
            float m = fmaxf(fmaxf(c[0][vv], c[1][vv]), fmaxf(c[2][vv], c[3][vv]));
            m = fmaxf(m, __shfl_xor(m, 1, 32));
            m = fmaxf(m, __shfl_xor(m, 2, 32));
            m = fmaxf(m, __shfl_xor(m, 4, 32));
            m = fmaxf(m, __shfl_xor(m, 8, 32));
            float ssum = 0.0f;
#pragma unroll
            for (int j = 0; j < 4; ++j) {
                float e = __expf((c[j][vv] - m) * scale);
                c[j][vv] = e;
                ssum += e;
            }
            ssum += __shfl_xor(ssum, 1, 32);
            ssum += __shfl_xor(ssum, 2, 32);
            ssum += __shfl_xor(ssum, 4, 32);
            ssum += __shfl_xor(ssum, 8, 32);
            const int M = vv + half_ * 8;
            const float coef = w[(l * HQn + h * Gg + M) * SBLK + s] / ssum;
#pragma unroll
            for (int j = 0; j < 4; ++j)
                psw[M * BSZ + 16 * j + l16] = (_Float16)(c[j][vv] * coef);
        }

        // ---- O += P x V : P as A-fragments from LDS, V as B-fragments ----
#pragma unroll
        for (int tp = 0; tp < 2; ++tp) {
            // A layout: element e -> u = 32tp + half*16 + e
            const v8h* pl = reinterpret_cast<const v8h*>(
                psw + l16 * BSZ + tp * 32 + half_ * 16);
            v8h lo = pl[0];
            v8h hi = pl[1];   // +8 halves
            v16h pa;
#pragma unroll
            for (int i = 0; i < 8; ++i) { pa[i] = lo[i]; pa[8 + i] = hi[i]; }

#pragma unroll
            for (int jd = 0; jd < 8; ++jd) {
                // B layout: lane -> N = d = 16jd + l16; element e -> u = 32tp + half*16 + e
                const float* vcol = v + kvbase + jd * 16 + l16;
                v16h vb;
#pragma unroll
                for (int e = 0; e < 16; ++e) {
                    const int u = tp * 32 + half_ * 16 + e;
                    vb[e] = (_Float16)vcol[(long)u * (HKV * DH)];
                }
                od[jd] = __builtin_amdgcn_wmma_f32_16x16x32_f16(
                    false, pa, false, vb, (short)0, od[jd], false, false);
            }
        }
    }

    // ---- write per-wave partial O tile to LDS, reduce across waves ----
#pragma unroll
    for (int jd = 0; jd < 8; ++jd)
#pragma unroll
        for (int vv = 0; vv < 8; ++vv)
            op[wave][(vv + half_ * 8) * DH + jd * 16 + l16] = od[jd][vv];
    __syncthreads();

    const int base = tid * 16;    // 128 threads x 16 floats = 2048
    float* obase = out + qbase;   // output tile is laid out like the Q tile
#pragma unroll
    for (int c4 = 0; c4 < 4; ++c4) {
        v4f r = {0.0f, 0.0f, 0.0f, 0.0f};
#pragma unroll
        for (int wv = 0; wv < 4; ++wv)
            r += *reinterpret_cast<const v4f*>(&op[wv][base + c4 * 4]);
        *reinterpret_cast<v4f*>(obase + base + c4 * 4) = r;
    }
}

extern "C" void kernel_launch(void* const* d_in, const int* in_sizes, int n_in,
                              void* d_out, int out_size, void* d_ws, size_t ws_size,
                              hipStream_t stream) {
    (void)in_sizes; (void)n_in; (void)out_size; (void)d_ws; (void)ws_size;
    const float* q  = (const float*)d_in[0];
    const float* k  = (const float*)d_in[1];
    const float* v  = (const float*)d_in[2];
    const float* w  = (const float*)d_in[3];
    const int*   bi = (const int*)d_in[4];
    float* out = (float*)d_out;

    dim3 grid(LQ * HKV);   // one workgroup per (query row, kv head)
    dim3 block(128);       // 4 waves (wave32)
    sparse_blk_attn_kernel<<<grid, block, 0, stream>>>(q, k, v, w, bi, out);
}
